// QuantizedStateSpaceMixer_36335423324365
// MI455X (gfx1250) — 16-bit and 8-bit matrix operand order repaired and checked by reading, compiled, not run on hardware
//
#include <hip/hip_runtime.h>
#include <math.h>

#define B_   2
#define T_   1024
#define DM_  1024
#define DI_  2048
#define DC_  4
#define DS_  16
#define DTR_ 128
#define T1_  (T_ + 1)
#define XDW_ (DTR_ + 2 * DS_)
#define QB_  127.0f

typedef __attribute__((ext_vector_type(16))) _Float16 v16h;
typedef __attribute__((ext_vector_type(8)))  float    v8f;
typedef __attribute__((ext_vector_type(8)))  int      v8i;

#if defined(__has_builtin)
#if __has_builtin(__builtin_amdgcn_global_load_async_to_lds_b128) && \
    __has_builtin(__builtin_amdgcn_s_wait_asynccnt)
#define HAS_ASYNC_LDS 1
#endif
#endif
#ifndef HAS_ASYNC_LDS
#define HAS_ASYNC_LDS 0
#endif

#if HAS_ASYNC_LDS
typedef int aint4 __attribute__((vector_size(16)));
typedef __attribute__((address_space(1))) aint4 g_aint4;
typedef __attribute__((address_space(3))) aint4 l_aint4;
__device__ __forceinline__ void async_copy16(const void* gsrc, void* ldst) {
  __builtin_amdgcn_global_load_async_to_lds_b128((g_aint4*)gsrc, (l_aint4*)ldst, 0, 0);
}
#endif

__global__ void k_zero(float* p, int n) {
  int i = blockIdx.x * blockDim.x + threadIdx.x;
  if (i < n) p[i] = 0.f;
}

__global__ void k_absmean(const float* __restrict__ w, int n, float* __restrict__ acc) {
  __shared__ float red[256];
  float s = 0.f;
  for (size_t i = (size_t)blockIdx.x * 256 + threadIdx.x; i < (size_t)n;
       i += (size_t)gridDim.x * 256)
    s += fabsf(w[i]);
  red[threadIdx.x] = s; __syncthreads();
  for (int st = 128; st > 0; st >>= 1) {
    if (threadIdx.x < st) red[threadIdx.x] += red[threadIdx.x + st];
    __syncthreads();
  }
  if (threadIdx.x == 0) atomicAdd(acc, red[0]);
}

__global__ void k_quant_w(const float* __restrict__ w, signed char* __restrict__ q,
                          const float* __restrict__ acc, float invN, size_t n) {
  size_t i = (size_t)blockIdx.x * blockDim.x + threadIdx.x;
  if (i >= n) return;
  float ws = fmaxf(acc[0] * invN, 1e-5f);
  float v = rintf(w[i] / ws);
  v = fminf(fmaxf(v, -1.f), 1.f);
  q[i] = (signed char)v;
}

__global__ void k_cvt_f16(const float* __restrict__ s, _Float16* __restrict__ d, size_t n) {
  size_t i = (size_t)blockIdx.x * blockDim.x + threadIdx.x;
  if (i < n) d[i] = (_Float16)s[i];
}

__global__ void k_rmsnorm_quant(const float* __restrict__ x, const float* __restrict__ nw,
                                signed char* __restrict__ q, float* __restrict__ rowscale) {
  __shared__ float red[256];
  const int row = blockIdx.x;
  const int tid = threadIdx.x;
  const float* xr = x + (size_t)row * DM_;
  float ss = 0.f;
  for (int i = tid; i < DM_; i += 256) { float v = xr[i]; ss += v * v; }
  red[tid] = ss; __syncthreads();
  for (int s = 128; s > 0; s >>= 1) {
    if (tid < s) red[tid] += red[tid + s];
    __syncthreads();
  }
  const float rs = rsqrtf(red[0] / (float)DM_ + 1e-5f);
  __syncthreads();
  float amax = 0.f, hv[DM_ / 256];
  int j = 0;
  for (int i = tid; i < DM_; i += 256, ++j) {
    float h = xr[i] * rs * nw[i];
    hv[j] = h;
    amax = fmaxf(amax, fabsf(h));
  }
  red[tid] = amax; __syncthreads();
  for (int s = 128; s > 0; s >>= 1) {
    if (tid < s) red[tid] = fmaxf(red[tid], red[tid + s]);
    __syncthreads();
  }
  amax = fmaxf(red[0], 1e-5f);
  const float xs = QB_ / amax;
  if (tid == 0) rowscale[row] = amax / QB_;
  j = 0;
  for (int i = tid; i < DM_; i += 256, ++j) {
    float qq = fminf(fmaxf(rintf(hv[j] * xs), -QB_), QB_);
    q[(size_t)row * DM_ + i] = (signed char)qq;
  }
}

__global__ void k_silu_quant(const float* __restrict__ x, signed char* __restrict__ q,
                             float* __restrict__ rowscale, int cols) {
  __shared__ float red[256];
  const int row = blockIdx.x;
  const int tid = threadIdx.x;
  const float* xr = x + (size_t)row * cols;
  float amax = 0.f, hv[DI_ / 256];
  int j = 0;
  for (int i = tid; i < cols; i += 256, ++j) {
    float v = xr[i];
    float h = v / (1.f + __expf(-v));
    hv[j] = h;
    amax = fmaxf(amax, fabsf(h));
  }
  red[tid] = amax; __syncthreads();
  for (int s = 128; s > 0; s >>= 1) {
    if (tid < s) red[tid] = fmaxf(red[tid], red[tid + s]);
    __syncthreads();
  }
  amax = fmaxf(red[0], 1e-5f);
  const float xs = QB_ / amax;
  if (tid == 0) rowscale[row] = amax / QB_;
  j = 0;
  for (int i = tid; i < cols; i += 256, ++j) {
    float qq = fminf(fmaxf(rintf(hv[j] * xs), -QB_), QB_);
    q[(size_t)row * cols + i] = (signed char)qq;
  }
}

__global__ void __launch_bounds__(256)
k_gemm_i8(const signed char* __restrict__ A, int K,
          const signed char* __restrict__ W, float* __restrict__ C,
          int M, int N, const float* __restrict__ rowscale,
          const float* __restrict__ wsum, float invN) {
  __shared__ __align__(16) signed char smB[2][64 * 64];
  const int lane = threadIdx.x, wave = threadIdx.y;
  const int ft = (wave << 5) + lane;
  const int tilesN = (N + 63) >> 6;
  const int n0 = (blockIdx.x % tilesN) << 6;
  const int m0 = ((blockIdx.x / tilesN) << 7) + (wave << 4);
  const int half = lane >> 4, l15 = lane & 15;
  const int m = m0 + l15;
  const int mc = (m < M) ? m : (M - 1);
  const size_t arow = (size_t)mc * K;

  const int sr = ft >> 2, sc = ft & 3;
  const int srow = (n0 + sr < N) ? (n0 + sr) : (N - 1);
  const signed char* gsrc = W + (size_t)srow * K + (sc << 4);
  signed char* ldst[2] = { &smB[0][(sr << 6) + (sc << 4)],
                           &smB[1][(sr << 6) + (sc << 4)] };

  v8i acc[4] = {};
#if HAS_ASYNC_LDS
  async_copy16(gsrc, ldst[0]);
#else
  *(uint4*)ldst[0] = *(const uint4*)gsrc;
#endif
  int buf = 0;
  for (int kk = 0; kk < K; kk += 64, buf ^= 1) {
#if HAS_ASYNC_LDS
    __builtin_amdgcn_s_wait_asynccnt(0);
#endif
    __syncthreads();
    if (kk + 64 < K) {
#if HAS_ASYNC_LDS
      async_copy16(gsrc + kk + 64, ldst[buf ^ 1]);
#else
      *(uint4*)ldst[buf ^ 1] = *(const uint4*)(gsrc + kk + 64);
#endif
    }
    __builtin_prefetch(A + arow + kk + 64, 0, 1);
    v8i a;
#pragma unroll
    for (int v = 0; v < 8; ++v) {
      int k = kk + ((v >> 1) << 4) + (half << 3) + ((v & 1) << 2);
      a[v] = *(const int*)(A + arow + k);
    }
#pragma unroll
    for (int wsub = 0; wsub < 4; ++wsub) {
      const signed char* bp = &smB[buf][((wsub << 4) + l15) << 6];
      v8i b;
#pragma unroll
      for (int v = 0; v < 8; ++v)
        b[v] = *(const int*)(bp + ((v >> 1) << 4) + (half << 3) + ((v & 1) << 2));
      acc[wsub] = __builtin_amdgcn_wmma_i32_16x16x64_iu8(
          true, a, true, b, acc[wsub], false, false);
    }
  }
  const float wsc = wsum[0] * invN;
#pragma unroll
  for (int wsub = 0; wsub < 4; ++wsub)
#pragma unroll
    for (int r = 0; r < 8; ++r) {
      int mm = m0 + r + (half << 3);
      int nn = n0 + (wsub << 4) + l15;
      if (mm < M && nn < N)
        C[(size_t)mm * N + nn] = (float)acc[wsub][r] * rowscale[mm] * wsc;
    }
}

__global__ void __launch_bounds__(256)
k_gemm_f16(const float* __restrict__ A, int lda,
           const _Float16* __restrict__ W, float* __restrict__ C,
           int M, int N, int K, const float* __restrict__ bias, int mode) {
  __shared__ __align__(16) _Float16 smB[2][64 * 32];
  const int lane = threadIdx.x, wave = threadIdx.y;
  const int ft = (wave << 5) + lane;
  const int tilesN = (N + 63) >> 6;
  const int n0 = (blockIdx.x % tilesN) << 6;
  const int m0 = ((blockIdx.x / tilesN) << 7) + (wave << 4);
  const int half = lane >> 4, l15 = lane & 15;
  const int m = m0 + l15;
  const int mc = (m < M) ? m : (M - 1);
  const size_t arow = (size_t)mc * lda;

  const int sr = ft >> 2, sc = ft & 3;
  const int srow = (n0 + sr < N) ? (n0 + sr) : (N - 1);
  const _Float16* gsrc = W + (size_t)srow * K + (sc << 3);
  _Float16* ldst[2] = { &smB[0][(sr << 5) + (sc << 3)],
                        &smB[1][(sr << 5) + (sc << 3)] };

  v8f acc[4] = {};
#if HAS_ASYNC_LDS
  async_copy16(gsrc, ldst[0]);
#else
  *(uint4*)ldst[0] = *(const uint4*)gsrc;
#endif
  int buf = 0;
  for (int kk = 0; kk < K; kk += 32, buf ^= 1) {
#if HAS_ASYNC_LDS
    __builtin_amdgcn_s_wait_asynccnt(0);
#endif
    __syncthreads();
    if (kk + 32 < K) {
#if HAS_ASYNC_LDS
      async_copy16(gsrc + kk + 32, ldst[buf ^ 1]);
#else
      *(uint4*)ldst[buf ^ 1] = *(const uint4*)(gsrc + kk + 32);
#endif
    }
    __builtin_prefetch(A + arow + kk + 32, 0, 1);
    v16h a;
#pragma unroll
    for (int v = 0; v < 8; ++v) {
      int k = kk + (half << 4) + (v << 1);
      float2 p = *(const float2*)(A + arow + k);
      a[2 * v]     = (_Float16)p.x;
      a[2 * v + 1] = (_Float16)p.y;
    }
#pragma unroll
    for (int wsub = 0; wsub < 4; ++wsub) {
      const _Float16* bp = &smB[buf][(((wsub << 4) + l15) << 5) + (half << 4)];
      v16h b;
#pragma unroll
      for (int v = 0; v < 16; ++v) b[v] = bp[v];
      acc[wsub] = __builtin_amdgcn_wmma_f32_16x16x32_f16(
          false, a, false, b, (short)0, acc[wsub], false, false);
    }
  }
#pragma unroll
  for (int wsub = 0; wsub < 4; ++wsub)
#pragma unroll
    for (int r = 0; r < 8; ++r) {
      int mm = m0 + r + (half << 3);
      int nn = n0 + (wsub << 4) + l15;
      if (mm < M && nn < N) {
        float v = acc[wsub][r];
        if (mode == 1) {
          v += bias[nn];
          v = (v > 20.f) ? v : log1pf(__expf(v));
        }
        C[(size_t)mm * N + nn] = v;
      }
    }
}

__global__ void k_dwconv_same(const float* __restrict__ x, const float* __restrict__ w,
                              const float* __restrict__ bias, float* __restrict__ y) {
  size_t idx = (size_t)blockIdx.x * blockDim.x + threadIdx.x;
  const size_t total = (size_t)B_ * T1_ * DI_;
  if (idx >= total) return;
  int c = (int)(idx % DI_);
  int t = (int)((idx / DI_) % T1_);
  int b = (int)(idx / ((size_t)DI_ * T1_));
  float s = bias[c];
#pragma unroll
  for (int k = 0; k < DC_; ++k) {
    int ti = t - 2 + k;
    if (ti >= 0 && ti < T_) s += x[((size_t)b * T_ + ti) * DI_ + c] * w[c * DC_ + k];
  }
  y[idx] = s;
}

__global__ void k_mconv_silu(const float* __restrict__ xz, const float* __restrict__ w,
                             const float* __restrict__ bias, float* __restrict__ u) {
  size_t idx = (size_t)blockIdx.x * blockDim.x + threadIdx.x;
  const size_t total = (size_t)B_ * T1_ * DI_;
  if (idx >= total) return;
  int c = (int)(idx % DI_);
  int t = (int)((idx / DI_) % T1_);
  int b = (int)(idx / ((size_t)DI_ * T1_));
  float s = bias[c];
#pragma unroll
  for (int k = 0; k < DC_; ++k) {
    int ti = t - 3 + k;
    if (ti >= 0) s += xz[((size_t)b * T1_ + ti) * (2 * DI_) + c] * w[c * DC_ + k];
  }
  u[idx] = s / (1.f + __expf(-s));
}

__global__ void __launch_bounds__(256) k_scan(
    const float* __restrict__ dt, const float* __restrict__ u,
    const float* __restrict__ xdbl, const float* __restrict__ xz,
    const float* __restrict__ Alog, const float* __restrict__ Dp,
    float* __restrict__ y) {
  __shared__ float Bs[DS_], Cs[DS_];
  const int b = blockIdx.x >> 3;
  const int d = ((blockIdx.x & 7) << 8) + threadIdx.x;
  float Ar[DS_], h[DS_];
#pragma unroll
  for (int n = 0; n < DS_; ++n) {
    Ar[n] = -__expf(Alog[d * DS_ + n]);
    h[n] = 0.f;
  }
  const float dp = Dp[d];
  for (int t = 0; t < T1_; ++t) {
    const size_t rb = (size_t)b * T1_ + t;
    if (threadIdx.x < DS_)
      Bs[threadIdx.x] = xdbl[rb * XDW_ + DTR_ + threadIdx.x];
    else if (threadIdx.x < 2 * DS_)
      Cs[threadIdx.x - DS_] = xdbl[rb * XDW_ + DTR_ + DS_ + (threadIdx.x - DS_)];
    __syncthreads();
    const float dtv = dt[rb * DI_ + d];
    const float uv  = u[rb * DI_ + d];
    const float zv  = xz[rb * (2 * DI_) + DI_ + d];
    const float du  = dtv * uv;
    float accv = 0.f;
#pragma unroll
    for (int n = 0; n < DS_; ++n) {
      h[n] = __expf(dtv * Ar[n]) * h[n] + du * Bs[n];
      accv += h[n] * Cs[n];
    }
    const float sz = zv / (1.f + __expf(-zv));
    y[rb * DI_ + d] = (accv + uv * dp) * sz;
    __syncthreads();
  }
}

__global__ void k_final(const float* __restrict__ x, const float* __restrict__ o2,
                        float* __restrict__ out) {
  size_t idx = (size_t)blockIdx.x * blockDim.x + threadIdx.x;
  const size_t total = (size_t)B_ * T_ * DM_;
  if (idx >= total) return;
  int dm = (int)(idx % DM_);
  int t  = (int)((idx / DM_) % T_);
  int b  = (int)(idx / ((size_t)DM_ * T_));
  out[idx] = x[idx] + o2[((size_t)b * T1_ + t) * DM_ + dm];
}

extern "C" void kernel_launch(void* const* d_in, const int* in_sizes, int n_in,
                              void* d_out, int out_size, void* d_ws, size_t ws_size,
                              hipStream_t stream) {
  (void)in_sizes; (void)n_in; (void)out_size; (void)ws_size;
  const float* x          = (const float*)d_in[0];
  const float* norm_w     = (const float*)d_in[1];
  const float* in_proj_w  = (const float*)d_in[2];
  const float* conv_w     = (const float*)d_in[3];
  const float* conv_b     = (const float*)d_in[4];
  const float* m_in_w     = (const float*)d_in[5];
  const float* m_conv_w   = (const float*)d_in[6];
  const float* m_conv_b   = (const float*)d_in[7];
  const float* m_x_w      = (const float*)d_in[8];
  const float* m_dt_w     = (const float*)d_in[9];
  const float* m_dt_b     = (const float*)d_in[10];
  const float* m_A_log    = (const float*)d_in[11];
  const float* m_D        = (const float*)d_in[12];
  const float* m_out_w    = (const float*)d_in[13];
  const float* out_proj_w = (const float*)d_in[14];
  float* out = (float*)d_out;

  char* wsb = (char*)d_ws;
  size_t off = 0;
  auto alloc = [&](size_t bytes) -> void* {
    void* p = wsb + off;
    off = (off + bytes + 255) & ~(size_t)255;
    return p;
  };
  float*       acc      = (float*)alloc(256);
  signed char* wq_in    = (signed char*)alloc((size_t)DI_ * DM_);
  signed char* wq_out   = (signed char*)alloc((size_t)DM_ * DI_);
  signed char* xq       = (signed char*)alloc((size_t)B_ * T1_ * DI_);
  float*       rowscale = (float*)alloc((size_t)B_ * T1_ * 4);
  float*       h1       = (float*)alloc((size_t)B_ * T_ * DI_ * 4);
  float*       h2b      = (float*)alloc((size_t)B_ * T1_ * DI_ * 4);
  float*       xzb      = (float*)alloc((size_t)B_ * T1_ * 2 * DI_ * 4);
  float*       ub       = (float*)alloc((size_t)B_ * T1_ * DI_ * 4);
  float*       xdbl     = (float*)alloc((size_t)B_ * T1_ * XDW_ * 4);
  float*       dtb      = (float*)alloc((size_t)B_ * T1_ * DI_ * 4);
  float*       yb       = (float*)alloc((size_t)B_ * T1_ * DI_ * 4);
  float*       mo       = (float*)alloc((size_t)B_ * T1_ * DI_ * 4);
  float*       o2       = (float*)alloc((size_t)B_ * T1_ * DM_ * 4);
  _Float16*    w16_in   = (_Float16*)alloc((size_t)2 * DI_ * DI_ * 2);
  _Float16*    w16_x    = (_Float16*)alloc((size_t)XDW_ * DI_ * 2);
  _Float16*    w16_dt   = (_Float16*)alloc((size_t)DI_ * DTR_ * 2);
  _Float16*    w16_out  = (_Float16*)alloc((size_t)DI_ * DI_ * 2);

  const size_t nW = (size_t)DI_ * DM_;
  const float invN = 1.f / (float)nW;
  const dim3 gblk(32, 8);
  auto ggrid = [](int M, int N) {
    return dim3((unsigned)(((M + 127) / 128) * ((N + 63) / 64)));
  };

  k_zero<<<1, 32, 0, stream>>>(acc, 8);
  k_absmean<<<512, 256, 0, stream>>>(in_proj_w,  (int)nW, acc + 0);
  k_absmean<<<512, 256, 0, stream>>>(out_proj_w, (int)nW, acc + 1);
  k_quant_w<<<(unsigned)((nW + 255) / 256), 256, 0, stream>>>(in_proj_w,  wq_in,  acc + 0, invN, nW);
  k_quant_w<<<(unsigned)((nW + 255) / 256), 256, 0, stream>>>(out_proj_w, wq_out, acc + 1, invN, nW);

  size_t n;
  n = (size_t)2 * DI_ * DI_;  k_cvt_f16<<<(unsigned)((n + 255) / 256), 256, 0, stream>>>(m_in_w,  w16_in,  n);
  n = (size_t)XDW_ * DI_;     k_cvt_f16<<<(unsigned)((n + 255) / 256), 256, 0, stream>>>(m_x_w,   w16_x,   n);
  n = (size_t)DI_ * DTR_;     k_cvt_f16<<<(unsigned)((n + 255) / 256), 256, 0, stream>>>(m_dt_w,  w16_dt,  n);
  n = (size_t)DI_ * DI_;      k_cvt_f16<<<(unsigned)((n + 255) / 256), 256, 0, stream>>>(m_out_w, w16_out, n);

  k_rmsnorm_quant<<<B_ * T_, 256, 0, stream>>>(x, norm_w, xq, rowscale);
  k_gemm_i8<<<ggrid(B_ * T_, DI_), gblk, 0, stream>>>(
      xq, DM_, wq_in, h1, B_ * T_, DI_, rowscale, acc + 0, invN);

  size_t nel = (size_t)B_ * T1_ * DI_;
  k_dwconv_same<<<(unsigned)((nel + 255) / 256), 256, 0, stream>>>(h1, conv_w, conv_b, h2b);

  k_gemm_f16<<<ggrid(B_ * T1_, 2 * DI_), gblk, 0, stream>>>(
      h2b, DI_, w16_in, xzb, B_ * T1_, 2 * DI_, DI_, nullptr, 0);
  k_mconv_silu<<<(unsigned)((nel + 255) / 256), 256, 0, stream>>>(xzb, m_conv_w, m_conv_b, ub);

  k_gemm_f16<<<ggrid(B_ * T1_, XDW_), gblk, 0, stream>>>(
      ub, DI_, w16_x, xdbl, B_ * T1_, XDW_, DI_, nullptr, 0);
  k_gemm_f16<<<ggrid(B_ * T1_, DI_), gblk, 0, stream>>>(
      xdbl, XDW_, w16_dt, dtb, B_ * T1_, DI_, DTR_, m_dt_b, 1);

  k_scan<<<16, 256, 0, stream>>>(dtb, ub, xdbl, xzb, m_A_log, m_D, yb);

  k_gemm_f16<<<ggrid(B_ * T1_, DI_), gblk, 0, stream>>>(
      yb, DI_, w16_out, mo, B_ * T1_, DI_, DI_, nullptr, 0);

  k_silu_quant<<<B_ * T1_, 256, 0, stream>>>(mo, xq, rowscale, DI_);
  k_gemm_i8<<<ggrid(B_ * T1_, DM_), gblk, 0, stream>>>(
      xq, DI_, wq_out, o2, B_ * T1_, DM_, rowscale, acc + 1, invN);

  size_t nout = (size_t)B_ * T_ * DM_;
  k_final<<<(unsigned)((nout + 255) / 256), 256, 0, stream>>>(x, o2, out);
}
